// PrimitiveCNO_30966714204220
// MI455X (gfx1250) — compile-verified, matrix operand order checked by reading, not run on hardware
//
#include <hip/hip_runtime.h>
#include <math.h>

#define Bsz 32
#define Nsz 4096
#define Csz 64
#define Hsz 256
#define Psz 8
#define Ksz 2

typedef __attribute__((ext_vector_type(16))) __bf16       v16bf;
typedef __attribute__((ext_vector_type(8)))  float        v8f;
typedef __attribute__((ext_vector_type(4)))  unsigned int v4u;
typedef __attribute__((ext_vector_type(2)))  __bf16       v2bf;
typedef __attribute__((ext_vector_type(2)))  float        v2f;

union Frag {
    v16bf bf;
    v4u   q[2];
    unsigned int w[8];
};

__device__ __forceinline__ unsigned int packbf2(float lo, float hi) {
#if __has_builtin(__builtin_amdgcn_cvt_pk_bf16_f32)
    auto r = __builtin_amdgcn_cvt_pk_bf16_f32(lo, hi);
    unsigned int u;
    __builtin_memcpy(&u, &r, 4);
    return u;
#else
    v2f f;
    f.x = lo; f.y = hi;
    v2bf b = __builtin_convertvector(f, v2bf);
    unsigned int u;
    __builtin_memcpy(&u, &b, 4);
    return u;
#endif
}

// Branchless erf (Abramowitz-Stegun 7.1.26, |err| ~ 1.5e-7) on native
// v_rcp_f32 / v_exp_f32 -- no EXEC-diverging regions between WMMAs.
__device__ __forceinline__ float erf_branchless(float x) {
    float ax = fabsf(x);
    float t  = __builtin_amdgcn_rcpf(fmaf(0.3275911f, ax, 1.0f));
    float p  = t * fmaf(t, fmaf(t, fmaf(t, fmaf(t, 1.061405429f,
                    -1.453152027f), 1.421413741f), -0.284496736f),
                    0.254829592f);
    float e  = __builtin_amdgcn_exp2f(-1.4426950408889634f * x * x);
    return copysignf(fmaf(-p, e, 1.0f), x);
}
__device__ __forceinline__ float gelu_erf(float x) {
    return 0.5f * x * (1.0f + erf_branchless(x * 0.70710678118654752f));
}

// K offset inside a 32-wide chunk for bf16 A/B fragments:
// VGPR v in 0..7 holds a packed pair; lane bit4 (hf) selects the 8-wide slice.
__device__ __forceinline__ int koff(int v, int hf) {
    return ((v & 4) << 2) + hf * 8 + ((v & 3) << 1);
}

// ---------------- Weight pre-pack: f32 -> bf16 in B-fragment dword order ---
// W1p dword index: (((e*16 + j)*2 + kc)*32 + lane)*8 + v   (j = H-tile)
// W2p dword index: (((e*8 + kc)*4 + nt)*32 + lane)*8 + v   (nt = C-tile)
__global__ void __launch_bounds__(256)
pack_weights(const float* __restrict__ W1, const float* __restrict__ W2,
             unsigned int* __restrict__ W1p, unsigned int* __restrict__ W2p) {
    const int idx = blockIdx.x * 256 + threadIdx.x;      // 0 .. 65535
    {
        const int v = idx & 7, lane = (idx >> 3) & 31;
        const int kc = (idx >> 8) & 1, j = (idx >> 9) & 15, e = idx >> 13;
        const int hf = lane >> 4, lm = lane & 15;
        const int k = kc * 32 + koff(v, hf);
        const int n = j * 16 + lm;
        const float* p = W1 + ((size_t)e * Csz + k) * Hsz + n;
        W1p[idx] = packbf2(p[0], p[Hsz]);                // (k, n), (k+1, n)
    }
    {
        const int v = idx & 7, lane = (idx >> 3) & 31;
        const int nt = (idx >> 8) & 3, kc = (idx >> 10) & 7, e = idx >> 13;
        const int hf = lane >> 4, lm = lane & 15;
        const int k = kc * 32 + koff(v, hf);
        const int n = nt * 16 + lm;
        const float* p = W2 + ((size_t)e * Hsz + k) * Csz + n;
        W2p[idx] = packbf2(p[0], p[Csz]);
    }
}

// ---------------- Router: mean-pool -> logits -> top-2 -> softmax ----------
__global__ void __launch_bounds__(256)
router_kernel(const float* __restrict__ u, const float* __restrict__ Wr,
              const float* __restrict__ br, int* __restrict__ eidx,
              float* __restrict__ ewt) {
    __shared__ float part[256];
    __shared__ float pooled[Csz];
    __shared__ float logits[Psz];
    const int b = blockIdx.x;
    const int t = threadIdx.x;
    const int c = t & (Csz - 1);
    const int seg = t >> 6;
    float s = 0.0f;
    for (int n = seg; n < Nsz; n += 4)
        s += u[((size_t)b * Nsz + n) * Csz + c];
    part[t] = s;
    __syncthreads();
    if (t < Csz)
        pooled[t] = (part[t] + part[t + 64] + part[t + 128] + part[t + 192]) *
                    (1.0f / (float)Nsz);
    __syncthreads();
    if (t < Psz) {
        float acc = br[t];
        #pragma unroll
        for (int cc = 0; cc < Csz; ++cc) acc += pooled[cc] * Wr[cc * Psz + t];
        logits[t] = acc;
    }
    __syncthreads();
    if (t == 0) {
        int i0 = 0; float v0 = logits[0];
        for (int p = 1; p < Psz; ++p)
            if (logits[p] > v0) { v0 = logits[p]; i0 = p; }
        int i1 = (i0 == 0) ? 1 : 0; float v1 = logits[i1];
        for (int p = 0; p < Psz; ++p)
            if (p != i0 && logits[p] > v1) { v1 = logits[p]; i1 = p; }
        float e1 = __expf(v1 - v0);             // e0 == 1 after max-shift
        float inv = 1.0f / (1.0f + e1);
        eidx[b * 2 + 0] = i0; eidx[b * 2 + 1] = i1;
        ewt[b * 2 + 0] = inv; ewt[b * 2 + 1] = e1 * inv;
    }
}

// ---------------- Main: per-sample sparse 2-expert MLP via bf16 WMMA -------
__global__ void __launch_bounds__(128)
moe_kernel(const float* __restrict__ u, const unsigned int* __restrict__ W1p,
           const float* __restrict__ b1, const unsigned int* __restrict__ W2p,
           const float* __restrict__ b2, const int* __restrict__ eidx,
           const float* __restrict__ ewt, float* __restrict__ out) {
    constexpr int HP = Hsz + 8;                  // padded h1 row (bf16 elems)
    __shared__ __bf16 h1s[64 * HP];              // 64 rows x 256 bf16 (+pad)

    const int tid  = threadIdx.x;
    const int lane = tid & 31;
    const int wv   = tid >> 5;                   // wave 0..3, owns 16 rows
    const int hf   = lane >> 4;                  // K-half selector
    const int lm   = lane & 15;                  // M / N within tile
    const int b    = blockIdx.y;
    const int rowBase = blockIdx.x * 64 + wv * 16;

    // --- A fragments of u (16 rows x K=64) as two 16x32 bf16 chunks -------
    Frag A[2];
    {
        const float* up = u + ((size_t)b * Nsz + rowBase + lm) * Csz;
        #pragma unroll
        for (int kc = 0; kc < 2; ++kc)
            #pragma unroll
            for (int v = 0; v < 8; ++v) {
                int k = kc * 32 + koff(v, hf);
                A[kc].w[v] = packbf2(up[k], up[k + 1]);
            }
    }

    v8f oacc[4] = {};                            // weighted delta accum (16x64)

    for (int kk = 0; kk < Ksz; ++kk) {
        const int   e  = eidx[b * 2 + kk];
        const float wk = ewt[b * 2 + kk];

        // warm WGP$/L2 for the second GEMM's packed weights during GEMM1
        __builtin_prefetch(W2p + e * 8192 + tid * 64, 0, 1);

        // --- GEMM1: h1 = gelu(u @ W1e + b1e), bf16 rows staged in LDS -----
        #pragma unroll 2
        for (int j = 0; j < Hsz / 16; ++j) {
            v8f acc = {};
            #pragma unroll
            for (int kc = 0; kc < 2; ++kc) {
                Frag Bf;                          // pre-packed: 2 x b128 loads
                const v4u* bp = (const v4u*)
                    (W1p + ((((size_t)(e * 16 + j) * 2 + kc) * 32 + lane) << 3));
                Bf.q[0] = bp[0];
                Bf.q[1] = bp[1];
                acc = __builtin_amdgcn_wmma_f32_16x16x32_bf16(
                        false, A[kc].bf, false, Bf.bf, (short)0, acc,
                        false, false);
            }
            const int hcol = j * 16 + lm;
            const float bb = b1[e * Hsz + hcol];
            #pragma unroll
            for (int r = 0; r < 8; ++r) {
                int mr = wv * 16 + hf * 8 + r;   // D layout: lane bit4 -> M+8
                h1s[mr * HP + hcol] = (__bf16)gelu_erf(acc[r] + bb);
            }
        }
        // h1 rows are produced & consumed by the same wave: DS-counter
        // ordering only, no cross-wave barrier required.

        // --- GEMM2: delta = h1 @ W2e + b2e ; oacc += wk * delta -----------
        v8f acc2[4] = {};
        #pragma unroll 2
        for (int kc = 0; kc < Hsz / 32; ++kc) {
            Frag Af;                                     // depends on kc only
            #pragma unroll
            for (int v = 0; v < 8; ++v) {
                int k = kc * 32 + koff(v, hf);
                Af.w[v] = *(const unsigned int*)
                          &h1s[(wv * 16 + lm) * HP + k];   // 4B aligned
            }
            #pragma unroll
            for (int nt = 0; nt < 4; ++nt) {
                Frag Bf;                          // pre-packed: 2 x b128 loads
                const v4u* bp = (const v4u*)
                    (W2p + ((((size_t)(e * 8 + kc) * 4 + nt) * 32 + lane) << 3));
                Bf.q[0] = bp[0];
                Bf.q[1] = bp[1];
                acc2[nt] = __builtin_amdgcn_wmma_f32_16x16x32_bf16(
                        false, Af.bf, false, Bf.bf, (short)0, acc2[nt],
                        false, false);
            }
        }
        #pragma unroll
        for (int nt = 0; nt < 4; ++nt) {
            const float bb = b2[e * Csz + nt * 16 + lm];
            #pragma unroll
            for (int r = 0; r < 8; ++r)
                oacc[nt][r] += wk * (acc2[nt][r] + bb);
        }
    }

    // --- out = u + sum_k w_k * delta_k -----------------------------------
    #pragma unroll
    for (int nt = 0; nt < 4; ++nt) {
        const int c = nt * 16 + lm;
        #pragma unroll
        for (int r = 0; r < 8; ++r) {
            const int row = rowBase + hf * 8 + r;
            const size_t off = ((size_t)b * Nsz + row) * Csz + c;
            out[off] = u[off] + oacc[nt][r];
        }
    }
}

extern "C" void kernel_launch(void* const* d_in, const int* in_sizes, int n_in,
                              void* d_out, int out_size, void* d_ws, size_t ws_size,
                              hipStream_t stream) {
    const float* u  = (const float*)d_in[0];
    const float* W1 = (const float*)d_in[1];
    const float* b1 = (const float*)d_in[2];
    const float* W2 = (const float*)d_in[3];
    const float* b2 = (const float*)d_in[4];
    const float* Wr = (const float*)d_in[5];
    const float* br = (const float*)d_in[6];
    float* out = (float*)d_out;

    // workspace layout: eidx (256B) | ewt (256B) | W1p (256KB) | W2p (256KB)
    char* ws = (char*)d_ws;
    int*          eidx = (int*)ws;
    float*        ewt  = (float*)(ws + 256);
    unsigned int* W1p  = (unsigned int*)(ws + 512);
    unsigned int* W2p  = (unsigned int*)(ws + 512 + (Psz * Csz * Hsz / 2) * 4);

    pack_weights<<<256, 256, 0, stream>>>(W1, W2, W1p, W2p);
    router_kernel<<<Bsz, 256, 0, stream>>>(u, Wr, br, eidx, ewt);
    moe_kernel<<<dim3(Nsz / 64, Bsz), 128, 0, stream>>>(u, W1p, b1, W2p, b2,
                                                        eidx, ewt, out);
}
